// Attention_8959301780095
// MI455X (gfx1250) — matrix operand order repaired and checked by reading, not compiled, not run on hardware
//
#include <hip/hip_runtime.h>
#include <hip/hip_bf16.h>

typedef _Float16 h16 __attribute__((ext_vector_type(16)));
typedef _Float16 h8  __attribute__((ext_vector_type(8)));
typedef float    f8  __attribute__((ext_vector_type(8)));

#define EXP2F(x) __builtin_amdgcn_exp2f(x)   // v_exp_f32 (log2-domain softmax)

// Problem constants (match reference)
constexpr int B_   = 8;
constexpr int S_   = 1024;
constexpr int H_   = 32;
constexpr int KVH_ = 8;
constexpr int D_   = 128;
constexpr int BS_  = 16;     // page (block) size
constexpr int BPB_ = 64;     // blocks per sequence
constexpr int NB_  = 512;    // total blocks
constexpr int T_   = B_ * S_;
constexpr int G_   = H_ / KVH_;   // 4 query heads per kv head
constexpr int QT_  = S_ / 16;     // q tiles per sequence
// 1/sqrt(D) * log2(e): scores live in log2 units so exp2 == hardware v_exp_f32
constexpr float SCALE2_ = 0.08838834764831845f * 1.4426950408889634f;

// ---- Workspace layout ----
// [0, 32KB)            : int inv[NB*BS]   (slot -> fresh token, or -1)
// [32KB, +16.8MB)      : _Float16 Kg[B][KVH][S][D]      (gathered, row-major)
// [+16.8MB, +33.6MB)   : _Float16 Vt[B][KVH][D][S]      (gathered, transposed)
constexpr size_t KV_HALF_ = (size_t)T_ * KVH_ * D_;   // 8.4M halves each

__global__ void inv_init(int* __restrict__ inv) {
  int i = blockIdx.x * blockDim.x + threadIdx.x;
  if (i < NB_ * BS_) inv[i] = -1;
}
__global__ void inv_build(const int* __restrict__ slot_mapping, int* __restrict__ inv) {
  int t = blockIdx.x * blockDim.x + threadIdx.x;
  if (t < T_) inv[slot_mapping[t]] = t;
}

// Resolve sequence position (b, t, kvh) to its source row (fresh overrides cache).
__device__ __forceinline__ const float* resolve_row(int b, int t, int kvh,
                                                    const float* __restrict__ fresh,
                                                    const float* __restrict__ cache,
                                                    const int* __restrict__ bt,
                                                    const int* __restrict__ inv) {
  int blk  = bt[b * BPB_ + (t >> 4)];
  int slot = blk * BS_ + (t & (BS_ - 1));
  int tt   = inv[slot];
  const float* base = (tt >= 0) ? fresh : cache;
  int row = (tt >= 0) ? tt : slot;
  return base + ((size_t)row * KVH_ + kvh) * (size_t)D_;
}

// ---- Gather K into f16 row-major [bh][t][d]; one thread per 8 elements ----
__global__ void gather_k(const float* __restrict__ k, const float* __restrict__ kc,
                         const int* __restrict__ bt, const int* __restrict__ inv,
                         _Float16* __restrict__ kg) {
  int gid = blockIdx.x * blockDim.x + threadIdx.x;
  if (gid >= (int)(KV_HALF_ / 8)) return;
  int d0  = (gid * 8) % D_;
  int lin = (gid * 8) / D_;          // (b*KVH + kvh)*S + t
  int t   = lin % S_;
  int bh  = lin / S_;
  const float* src = resolve_row(bh / KVH_, t, bh % KVH_, k, kc, bt, inv);
  const float4* s4 = (const float4*)(src + d0);
  float4 x0 = s4[0], x1 = s4[1];
  h8 o;
  o[0] = (_Float16)x0.x; o[1] = (_Float16)x0.y;
  o[2] = (_Float16)x0.z; o[3] = (_Float16)x0.w;
  o[4] = (_Float16)x1.x; o[5] = (_Float16)x1.y;
  o[6] = (_Float16)x1.z; o[7] = (_Float16)x1.w;
  *(h8*)(kg + (size_t)lin * D_ + d0) = o;
}

// ---- Gather V into f16 transposed [bh][d][t]; one wave per 32 tokens ----
__global__ __launch_bounds__(32)
void transpose_v(const float* __restrict__ v, const float* __restrict__ vc,
                 const int* __restrict__ bt, const int* __restrict__ inv,
                 _Float16* __restrict__ vt) {
  int lane = threadIdx.x;
  int wid  = blockIdx.x;
  int tt32 = wid % (S_ / 32);
  int bh   = wid / (S_ / 32);
  int t    = tt32 * 32 + lane;
  const float* src = resolve_row(bh / KVH_, t, bh % KVH_, v, vc, bt, inv);
  _Float16* dst = vt + (size_t)bh * D_ * S_ + t;
#pragma unroll
  for (int d4 = 0; d4 < D_ / 4; ++d4) {
    float4 x = ((const float4*)src)[d4];
    dst[(size_t)(d4 * 4 + 0) * S_] = (_Float16)x.x;   // lanes write 64B contiguous
    dst[(size_t)(d4 * 4 + 1) * S_] = (_Float16)x.y;
    dst[(size_t)(d4 * 4 + 2) * S_] = (_Float16)x.z;
    dst[(size_t)(d4 * 4 + 3) * S_] = (_Float16)x.w;
  }
}

// ---- One 32-key tile step of flash attention (template: causal mask on/off) ----
template <bool MASK>
__device__ __forceinline__ void tile_step(
    int kt, int q0, bool hi, int ln,
    const _Float16* __restrict__ kbh, const _Float16* __restrict__ vbh,
    const h16 (&qa)[4], f8 (&o)[8], float (&m_i)[8], float (&l_i)[8],
    _Float16* sP) {
  // ---- scores: S(16x32) = Q(16x128) @ K^T(128x32); B-tiles contiguous f16
  f8 sc[2];
#pragma unroll
  for (int nt = 0; nt < 2; ++nt) {
    const _Float16* kr =
        kbh + (size_t)(kt * 32 + nt * 16 + ln) * D_ + (hi ? 16 : 0);
    f8 acc = {};
#pragma unroll
    for (int c = 0; c < 4; ++c) {
      h16 bk = *(const h16*)(kr + c * 32);   // 32B aligned, 2x b128
      acc = __builtin_amdgcn_wmma_f32_16x16x32_f16(
          false, qa[c], false, bk, (short)0, acc, false, false);
    }
    sc[nt] = acc;
  }

  // ---- online softmax in log2 units (C layout: row = r + 8*hi, col = ln)
  float alpha[8];
#pragma unroll
  for (int r = 0; r < 8; ++r) {
    float s0 = sc[0][r], s1 = sc[1][r];
    if (MASK) {
      const int qp  = q0 + r + (hi ? 8 : 0);
      const int kp0 = kt * 32 + ln;
      s0 = (kp0      <= qp) ? s0 : -1e30f;
      s1 = (kp0 + 16 <= qp) ? s1 : -1e30f;
    }
    float tm = fmaxf(s0, s1);
#pragma unroll
    for (int mk = 1; mk < 16; mk <<= 1)
      tm = fmaxf(tm, __shfl_xor(tm, mk, 32));
    const float mn = fmaxf(m_i[r], tm);
    alpha[r] = EXP2F(m_i[r] - mn);
    m_i[r] = mn;
    sc[0][r] = EXP2F(s0 - mn);
    sc[1][r] = EXP2F(s1 - mn);
  }
#pragma unroll
  for (int db = 0; db < 8; ++db)
#pragma unroll
    for (int r = 0; r < 8; ++r) o[db][r] *= alpha[r];

  // ---- P: C layout -> A layout via 1KB LDS bounce (f16)
#pragma unroll
  for (int r = 0; r < 8; ++r) {
    const int m = r + (hi ? 8 : 0);
    sP[m * 32 + ln]      = (_Float16)sc[0][r];
    sP[m * 32 + 16 + ln] = (_Float16)sc[1][r];
  }
  asm volatile("s_wait_dscnt 0" ::: "memory");   // cross-lane LDS RAW
  h16 pa;
  {
    const int kb = hi ? 16 : 0;
    h8 g0 = *(const h8*)(sP + ln * 32 + kb);
    h8 g1 = *(const h8*)(sP + ln * 32 + kb + 8);
#pragma unroll
    for (int i = 0; i < 8; ++i) { pa[i] = g0[i]; pa[8 + i] = g1[i]; }
  }

  // ---- row-sum of P via WMMA against all-ones B-tile: lands in C layout,
  //      same slot as l_i[r] for every lane -> no cross-lane reduction needed
  {
    h16 ones;
#pragma unroll
    for (int i = 0; i < 16; ++i) ones[i] = (_Float16)1.0f;
    f8 z = {};
    f8 ls = __builtin_amdgcn_wmma_f32_16x16x32_f16(
        false, pa, false, ones, (short)0, z, false, false);
#pragma unroll
    for (int r = 0; r < 8; ++r) l_i[r] = l_i[r] * alpha[r] + ls[r];
  }

  // ---- PV: O(16x128) += P(16x32) @ V(32x128); V B-tiles contiguous in Vt
#pragma unroll
  for (int db = 0; db < 8; ++db) {
    const _Float16* vr =
        vbh + (size_t)(db * 16 + ln) * S_ + kt * 32 + (hi ? 16 : 0);
    h16 bv = *(const h16*)(vr);                // 32B aligned, 2x b128
    o[db] = __builtin_amdgcn_wmma_f32_16x16x32_f16(
        false, pa, false, bv, (short)0, o[db], false, false);
  }
}

// ---- Flash attention: one wave = 16-query x 128 output tile of one head ----
__global__ __launch_bounds__(32, 4)
void fa_kernel(const float* __restrict__ q,
               const _Float16* __restrict__ kg,
               const _Float16* __restrict__ vt,
               float* __restrict__ out) {
  __shared__ _Float16 sP[16 * 32] __attribute__((aligned(16)));

  const int lane = threadIdx.x;
  const bool hi  = lane >= 16;
  const int  ln  = lane & 15;

  const int gid = blockIdx.x;
  const int qt  = gid % QT_;
  const int h   = (gid / QT_) % H_;
  const int b   = gid / (QT_ * H_);
  const int kh  = h / G_;
  const int q0  = qt * 16;

  const _Float16* kbh = kg + (size_t)(b * KVH_ + kh) * S_ * D_;
  const _Float16* vbh = vt + (size_t)(b * KVH_ + kh) * D_ * S_;

  // ---- Q tile in WMMA A layout (M = lane%16, K halves {kb..kb+15})
  h16 qa[4];
  {
    const float* qrow = q + (size_t)(b * S_ + q0 + ln) * (H_ * D_) + h * D_;
    const int kb = hi ? 16 : 0;
#pragma unroll
    for (int c = 0; c < 4; ++c) {
      const float4* s0 = (const float4*)(qrow + c * 32 + kb);
      const float4* s1 = (const float4*)(qrow + c * 32 + kb + 8);
      float4 a0 = s0[0], a1 = s0[1], b0 = s1[0], b1 = s1[1];
      qa[c][0]  = (_Float16)(a0.x * SCALE2_); qa[c][1]  = (_Float16)(a0.y * SCALE2_);
      qa[c][2]  = (_Float16)(a0.z * SCALE2_); qa[c][3]  = (_Float16)(a0.w * SCALE2_);
      qa[c][4]  = (_Float16)(a1.x * SCALE2_); qa[c][5]  = (_Float16)(a1.y * SCALE2_);
      qa[c][6]  = (_Float16)(a1.z * SCALE2_); qa[c][7]  = (_Float16)(a1.w * SCALE2_);
      qa[c][8]  = (_Float16)(b0.x * SCALE2_); qa[c][9]  = (_Float16)(b0.y * SCALE2_);
      qa[c][10] = (_Float16)(b0.z * SCALE2_); qa[c][11] = (_Float16)(b0.w * SCALE2_);
      qa[c][12] = (_Float16)(b1.x * SCALE2_); qa[c][13] = (_Float16)(b1.y * SCALE2_);
      qa[c][14] = (_Float16)(b1.z * SCALE2_); qa[c][15] = (_Float16)(b1.w * SCALE2_);
    }
  }

  f8 o[8];
#pragma unroll
  for (int i = 0; i < 8; ++i) o[i] = {};
  float m_i[8], l_i[8];
#pragma unroll
  for (int r = 0; r < 8; ++r) { m_i[r] = -1e30f; l_i[r] = 0.f; }

  // Tiles [0, q0/32) are fully causal-valid; exactly one masked tile follows.
  const int nfull = q0 >> 5;
  for (int kt = 0; kt < nfull; ++kt)
    tile_step<false>(kt, q0, hi, ln, kbh, vbh, qa, o, m_i, l_i, sP);
  tile_step<true>(nfull, q0, hi, ln, kbh, vbh, qa, o, m_i, l_i, sP);

  // ---- epilogue: normalize and store
  float rl[8];
#pragma unroll
  for (int r = 0; r < 8; ++r) rl[r] = 1.0f / l_i[r];
#pragma unroll
  for (int db = 0; db < 8; ++db) {
#pragma unroll
    for (int r = 0; r < 8; ++r) {
      const int m = r + (hi ? 8 : 0);
      out[(size_t)(b * S_ + q0 + m) * (H_ * D_) + h * D_ + db * 16 + ln] =
          o[db][r] * rl[r];
    }
  }
}

extern "C" void kernel_launch(void* const* d_in, const int* in_sizes, int n_in,
                              void* d_out, int out_size, void* d_ws, size_t ws_size,
                              hipStream_t stream) {
  (void)in_sizes; (void)n_in; (void)out_size; (void)ws_size;
  const float* q  = (const float*)d_in[0];
  const float* k  = (const float*)d_in[1];
  const float* v  = (const float*)d_in[2];
  const float* kc = (const float*)d_in[3];
  const float* vc = (const float*)d_in[4];
  const int* slot_mapping = (const int*)d_in[5];
  const int* bt   = (const int*)d_in[6];
  float* out = (float*)d_out;

  int* inv      = (int*)d_ws;
  _Float16* kg  = (_Float16*)((char*)d_ws + 32768);
  _Float16* vtb = kg + KV_HALF_;

  inv_init<<<(NB_ * BS_ + 255) / 256, 256, 0, stream>>>(inv);
  inv_build<<<(T_ + 255) / 256, 256, 0, stream>>>(slot_mapping, inv);
  gather_k<<<((int)(KV_HALF_ / 8) + 255) / 256, 256, 0, stream>>>(k, kc, bt, inv, kg);
  transpose_v<<<B_ * KVH_ * (S_ / 32), 32, 0, stream>>>(v, vc, bt, inv, vtb);
  fa_kernel<<<B_ * H_ * QT_, 32, 0, stream>>>(q, kg, vtb, out);
}
